// PunctCapBiLstmCrf_36799279792561
// MI455X (gfx1250) — matrix operand order repaired and checked by reading, not compiled, not run on hardware
//
#include <hip/hip_runtime.h>
#include <hip/hip_bf16.h>

// ---------------------------------------------------------------------------
// PunctCap BiLSTM-CRF for MI455X (gfx1250, wave32, WMMA).
//
// Plan (see analysis): bf16 WMMA for the hoisted input projections and the
// per-step recurrent GEMM; h in LDS (bf16), c in VGPRs; gate pre-activations
// streamed as bf16 (~50MB) — trivial vs 23.3 TB/s HBM; weights (~14MB bf16)
// stay resident in the 192MB L2.
// ---------------------------------------------------------------------------

constexpr int kB  = 32;
constexpr int kL  = 256;
constexpr int kD  = 768;   // model width / lstm input width (both layers)
constexpr int kH  = 384;   // hidden per direction
constexpr int kG  = 4 * kH; // 1536 gate width
constexpr int kNP = 9;
constexpr int kNC = 3;

typedef __attribute__((ext_vector_type(16))) __bf16        v16bf;
typedef __attribute__((ext_vector_type(8)))  float         v8f;
typedef __attribute__((ext_vector_type(4)))  unsigned int  uint32x4;

union BFrag { v16bf v; uint32x4 q[2]; };

__device__ inline v8f v8f_zero() {
    v8f z; for (int i = 0; i < 8; ++i) z[i] = 0.f; return z;
}
__device__ inline unsigned short f2bu(float x) {       // f32 -> bf16 (RNE)
    unsigned u = __float_as_uint(x);
    u += 0x7FFFu + ((u >> 16) & 1u);
    return (unsigned short)(u >> 16);
}
__device__ inline float bu2f(unsigned short u) {       // bf16 -> f32
    return __uint_as_float(((unsigned)u) << 16);
}
__device__ inline float sigm(float x) { return 1.f / (1.f + expf(-x)); }

// ---------------------------------------------------------------------------
// Elementwise conversion / bias helpers
// ---------------------------------------------------------------------------
__global__ void f32_to_bf16_kernel(const float* __restrict__ in,
                                   unsigned short* __restrict__ out, int n) {
    int i = blockIdx.x * blockDim.x + threadIdx.x;
    if (i < n) out[i] = f2bu(in[i]);
}

__global__ void bias_sum_kernel(const float* __restrict__ a,
                                const float* __restrict__ b,
                                float* __restrict__ o, int n) {
    int i = blockIdx.x * blockDim.x + threadIdx.x;
    if (i < n) o[i] = a[i] + b[i];
}

// ---------------------------------------------------------------------------
// Embedding gather + LayerNorm -> bf16 activations [B*L][D]
// one block per token, 256 threads, 3 elems/thread
// ---------------------------------------------------------------------------
__global__ void __launch_bounds__(256)
embed_ln_kernel(const int* __restrict__ ids, const int* __restrict__ tt,
                const float* __restrict__ wemb, const float* __restrict__ pemb,
                const float* __restrict__ temb, const float* __restrict__ g,
                const float* __restrict__ bta, unsigned short* __restrict__ xbf) {
    const int row = blockIdx.x;          // b*L + l
    const int l   = row & (kL - 1);
    const int tid = threadIdx.x;
    const float* wr = wemb + (size_t)ids[row] * kD;
    const float* pr = pemb + (size_t)l * kD;
    const float* tr = temb + (size_t)tt[row] * kD;

    float v[3]; float s = 0.f;
    for (int i = 0; i < 3; ++i) {
        int d = tid + i * 256;
        v[i] = wr[d] + pr[d] + tr[d];
        s += v[i];
    }
    __shared__ float red[256];
    red[tid] = s; __syncthreads();
    for (int off = 128; off > 0; off >>= 1) {
        if (tid < off) red[tid] += red[tid + off];
        __syncthreads();
    }
    const float mu = red[0] / (float)kD;
    __syncthreads();
    float vs = 0.f;
    for (int i = 0; i < 3; ++i) { float d = v[i] - mu; vs += d * d; }
    red[tid] = vs; __syncthreads();
    for (int off = 128; off > 0; off >>= 1) {
        if (tid < off) red[tid] += red[tid + off];
        __syncthreads();
    }
    const float rstd = rsqrtf(red[0] / (float)kD + 1e-12f);
    for (int i = 0; i < 3; ++i) {
        int d = tid + i * 256;
        xbf[(size_t)row * kD + d] = f2bu((v[i] - mu) * rstd * g[d] + bta[d]);
    }
}

// ---------------------------------------------------------------------------
// Hoisted input projection: gpre[dir][m][n] = x @ Wih_dir^T + (bih+bhh)
// bf16 WMMA 16x16x32, block tile 64M x 128N (8 waves, wave tile 16x64)
// grid: (8192/64, 1536/128, 2 dirs)
// ---------------------------------------------------------------------------
__global__ void __launch_bounds__(256)
gemm_gates_kernel(const unsigned short* __restrict__ xbf,  // [8192][768]
                  const unsigned short* __restrict__ Wih,  // [2][1536][768] (layer slice)
                  const float* __restrict__ bias,          // [2][1536]
                  unsigned short* __restrict__ gpre) {     // [2][8192][1536]
    const int dir = blockIdx.z;
    const unsigned short* W = Wih + (size_t)dir * kG * kD;
    const float* bs = bias + (size_t)dir * kG;
    unsigned short* out = gpre + (size_t)dir * (size_t)(kB * kL) * kG;

    const int lane = threadIdx.x & 31;
    const int wv   = threadIdx.x >> 5;              // 0..7
    const int m0   = blockIdx.x * 64 + (wv >> 1) * 16;
    const int n0   = blockIdx.y * 128 + (wv & 1) * 64;

    v8f acc[4];
    for (int i = 0; i < 4; ++i) acc[i] = v8f_zero();

    const int arow = m0 + (lane & 15);
    const int kA   = (lane >> 4) << 3;    // +8 for upper lane half (A layout)
    const int kBo  = (lane >> 4) << 3;    // +8 for upper lane half (B layout)

    for (int k0 = 0; k0 < kD; k0 += 32) {
        BFrag a;
        const unsigned short* pA = xbf + (size_t)arow * kD + k0 + kA;
        a.q[0] = *(const uint32x4*)(pA);
        a.q[1] = *(const uint32x4*)(pA + 16);
        __builtin_prefetch(pA + 32, 0, 1);          // global_prefetch_b8
        for (int i = 0; i < 4; ++i) {
            BFrag bf;
            const unsigned short* pB =
                W + (size_t)(n0 + 16 * i + (lane & 15)) * kD + k0 + kBo;
            bf.q[0] = *(const uint32x4*)(pB);
            bf.q[1] = *(const uint32x4*)(pB + 16);
            acc[i] = __builtin_amdgcn_wmma_f32_16x16x32_bf16(
                false, a.v, false, bf.v, (short)0, acc[i], false, false);
        }
    }
    const int moff = (lane >> 4) << 3;
    for (int i = 0; i < 4; ++i) {
        const int n = n0 + 16 * i + (lane & 15);
        const float bn = bs[n];
        for (int r = 0; r < 8; ++r) {
            const int m = m0 + r + moff;
            out[(size_t)m * kG + n] = f2bu(acc[i][r] + bn);
        }
    }
}

// ---------------------------------------------------------------------------
// Recurrent LSTM layer, persistent: grid(2 dirs), 768 threads = 24 waves.
// Wave wv owns hidden columns [wv*16, wv*16+16) for all 4 gates, both M tiles;
// c lives in VGPRs, h in LDS (bf16) + f32 copy streamed to hout.
// ---------------------------------------------------------------------------
__global__ void __launch_bounds__(768)
lstm_layer_kernel(const unsigned short* __restrict__ gpre,  // [2][8192][1536]
                  const unsigned short* __restrict__ Whh2,  // [2][1536][384]
                  float* __restrict__ hout) {               // [8192][768]
    const int dir = blockIdx.x;
    const unsigned short* W  = Whh2 + (size_t)dir * kG * kH;
    const unsigned short* gp = gpre + (size_t)dir * (size_t)(kB * kL) * kG;

    __shared__ unsigned short hbf[kB * kH];   // 32*384 bf16 = 24KB

    const int tid  = threadIdx.x;
    const int lane = tid & 31;
    const int wv   = tid >> 5;                // 0..23
    const int nb   = wv * 16;
    const int kA   = (lane >> 4) << 3;
    const int kBo  = (lane >> 4) << 3;
    const int moff = (lane >> 4) << 3;

    for (int i = tid; i < kB * kH; i += 768) hbf[i] = 0;
    float c0[8], c1[8];
    for (int r = 0; r < 8; ++r) { c0[r] = 0.f; c1[r] = 0.f; }
    __syncthreads();

    for (int step = 0; step < kL; ++step) {
        const int t = (dir == 0) ? step : (kL - 1 - step);

        v8f acc[4][2];
        for (int g2 = 0; g2 < 4; ++g2)
            for (int mt = 0; mt < 2; ++mt) acc[g2][mt] = v8f_zero();

        for (int k0 = 0; k0 < kH; k0 += 32) {
            BFrag a0, a1;
            {
                const unsigned short* p = &hbf[(lane & 15) * kH + k0 + kA];
                a0.q[0] = *(const uint32x4*)(p);
                a0.q[1] = *(const uint32x4*)(p + 16);
            }
            {
                const unsigned short* p = &hbf[(16 + (lane & 15)) * kH + k0 + kA];
                a1.q[0] = *(const uint32x4*)(p);
                a1.q[1] = *(const uint32x4*)(p + 16);
            }
            for (int g2 = 0; g2 < 4; ++g2) {
                BFrag bf;
                const unsigned short* p =
                    W + (size_t)(g2 * kH + nb + (lane & 15)) * kH + k0 + kBo;
                bf.q[0] = *(const uint32x4*)(p);
                bf.q[1] = *(const uint32x4*)(p + 16);
                acc[g2][0] = __builtin_amdgcn_wmma_f32_16x16x32_bf16(
                    false, a0.v, false, bf.v, (short)0, acc[g2][0], false, false);
                acc[g2][1] = __builtin_amdgcn_wmma_f32_16x16x32_bf16(
                    false, a1.v, false, bf.v, (short)0, acc[g2][1], false, false);
            }
        }
        __syncthreads();   // all reads of hbf for step t complete

        for (int mt = 0; mt < 2; ++mt) {
            float* cc = mt ? c1 : c0;
            for (int r = 0; r < 8; ++r) {
                const int m = mt * 16 + r + moff;        // batch index
                const int n = nb + (lane & 15);          // hidden index
                const size_t row = (size_t)m * kL + t;   // token row
                const unsigned short* gr = gp + row * kG;
                const float gi = acc[0][mt][r] + bu2f(gr[0 * kH + n]);
                const float gf = acc[1][mt][r] + bu2f(gr[1 * kH + n]);
                const float gg = acc[2][mt][r] + bu2f(gr[2 * kH + n]);
                const float go = acc[3][mt][r] + bu2f(gr[3 * kH + n]);
                const float cn = sigm(gf) * cc[r] + sigm(gi) * tanhf(gg);
                const float hn = sigm(go) * tanhf(cn);
                cc[r] = cn;
                hout[row * (size_t)kD + dir * kH + n] = hn;
                hbf[m * kH + n] = f2bu(hn);
            }
        }
        __syncthreads();   // hbf for step t+1 ready
    }
}

// ---------------------------------------------------------------------------
// Ragged compaction destinations (cumsum of valid_ids per batch)
// ---------------------------------------------------------------------------
__global__ void dest_kernel(const int* __restrict__ valid, int* __restrict__ dest) {
    const int b = threadIdx.x;
    if (b >= kB) return;
    int c = 0;
    for (int l = 0; l < kL; ++l) {
        if (valid[b * kL + l]) { dest[b * kL + l] = c; ++c; }
        else                     dest[b * kL + l] = -1;
    }
}

// Fill head logits with bias (rows never scattered-to stay at bias, per ref)
__global__ void init_heads_kernel(const float* __restrict__ pb,
                                  const float* __restrict__ cb,
                                  float* __restrict__ plog,
                                  float* __restrict__ clog) {
    const int row = blockIdx.x, t = threadIdx.x;
    if (t < kNP)             plog[(size_t)row * kNP + t] = pb[t];
    else if (t < kNP + kNC)  clog[(size_t)row * kNC + (t - kNP)] = cb[t - kNP];
}

// Head GEMV: 12 waves = 9 punct + 3 cap outputs; wave-level dot over 768.
__global__ void __launch_bounds__(384)
logits_kernel(const float* __restrict__ seq, const int* __restrict__ valid,
              const int* __restrict__ dest, const float* __restrict__ pW,
              const float* __restrict__ pb, const float* __restrict__ cW,
              const float* __restrict__ cb, float* __restrict__ plog,
              float* __restrict__ clog) {
    const int token = blockIdx.x;
    if (!valid[token]) return;
    const int b = token >> 8;
    const int r = dest[token];
    const size_t orow = (size_t)b * kL + r;
    const int wv = threadIdx.x >> 5, lane = threadIdx.x & 31;
    const bool isP = (wv < kNP);
    const float* wrow = isP ? (pW + (size_t)wv * kD)
                            : (cW + (size_t)(wv - kNP) * kD);
    const float* x = seq + (size_t)token * kD;
    float s = 0.f;
    for (int k = lane; k < kD; k += 32) s += x[k] * wrow[k];
    for (int off = 16; off > 0; off >>= 1) s += __shfl_xor(s, off, 32);
    if (lane == 0) {
        if (isP) plog[orow * kNP + wv] = s + pb[wv];
        else     clog[orow * kNC + (wv - kNP)] = s + cb[wv - kNP];
    }
}

// ---------------------------------------------------------------------------
// CRF: log-likelihood (forward algo) + Viterbi decode. T<=9, scalar per
// (batch, task) block. Deterministic per-block ll output (no atomics).
// ---------------------------------------------------------------------------
__global__ void crf_kernel(const float* __restrict__ plog, const float* __restrict__ clog,
                           const int* __restrict__ plabels, const int* __restrict__ clabels,
                           const int* __restrict__ maskI,
                           const float* __restrict__ p_start, const float* __restrict__ p_end,
                           const float* __restrict__ p_trans,
                           const float* __restrict__ c_start, const float* __restrict__ c_end,
                           const float* __restrict__ c_trans,
                           float* __restrict__ llbuf,      // [2][32]
                           float* __restrict__ out_tags,   // d_out + 3
                           int* __restrict__ bp) {         // [2][32][256][16]
    if (threadIdx.x != 0) return;
    const int b = blockIdx.x, task = blockIdx.y;
    const int T = (task == 0) ? kNP : kNC;
    const float* em = ((task == 0) ? plog : clog) + (size_t)b * kL * T;
    const int* tags = ((task == 0) ? plabels : clabels) + b * kL;
    const float* st = (task == 0) ? p_start : c_start;
    const float* en = (task == 0) ? p_end : c_end;
    const float* tr = (task == 0) ? p_trans : c_trans;
    const int* mk = maskI + b * kL;
    int* hist = bp + ((size_t)task * kB + b) * (size_t)kL * 16;
    float* outp = out_tags + (size_t)task * (kB * kL) + (size_t)b * kL;

    // ---- numerator ----
    int nvalid = 0;
    for (int l = 0; l < kL; ++l) nvalid += (mk[l] != 0);
    float num = st[tags[0]] + em[tags[0]];
    for (int l = 1; l < kL; ++l)
        if (mk[l]) num += tr[tags[l - 1] * T + tags[l]] + em[(size_t)l * T + tags[l]];
    num += en[tags[nvalid - 1]];

    // ---- forward (logZ) ----
    float sc[kNP], nx[kNP];
    for (int j = 0; j < T; ++j) sc[j] = st[j] + em[j];
    for (int l = 1; l < kL; ++l) {
        if (!mk[l]) continue;
        for (int cur = 0; cur < T; ++cur) {
            float mx = -3.4e38f;
            for (int pv = 0; pv < T; ++pv) {
                float v = sc[pv] + tr[pv * T + cur];
                if (v > mx) mx = v;
            }
            float ssum = 0.f;
            for (int pv = 0; pv < T; ++pv)
                ssum += expf(sc[pv] + tr[pv * T + cur] - mx);
            nx[cur] = mx + logf(ssum) + em[(size_t)l * T + cur];
        }
        for (int j = 0; j < T; ++j) sc[j] = nx[j];
    }
    float mx = -3.4e38f;
    for (int j = 0; j < T; ++j) { float v = sc[j] + en[j]; if (v > mx) mx = v; }
    float ssum = 0.f;
    for (int j = 0; j < T; ++j) ssum += expf(sc[j] + en[j] - mx);
    const float logZ = mx + logf(ssum);
    llbuf[task * kB + b] = num - logZ;

    // ---- Viterbi ----
    for (int j = 0; j < T; ++j) sc[j] = st[j] + em[j];
    for (int l = 1; l < kL; ++l) {
        if (mk[l]) {
            for (int cur = 0; cur < T; ++cur) {
                float best = -3.4e38f; int bi = 0;
                for (int pv = 0; pv < T; ++pv) {
                    float v = sc[pv] + tr[pv * T + cur];
                    if (v > best) { best = v; bi = pv; }
                }
                nx[cur] = best + em[(size_t)l * T + cur];
                hist[l * 16 + cur] = bi;
            }
            for (int j = 0; j < T; ++j) sc[j] = nx[j];
        } else {
            for (int cur = 0; cur < T; ++cur) hist[l * 16 + cur] = cur;
        }
    }
    int last = 0; { float best = -3.4e38f;
        for (int j = 0; j < T; ++j) { float v = sc[j] + en[j];
            if (v > best) { best = v; last = j; } } }
    int tcur = last;
    outp[kL - 1] = mk[kL - 1] ? (float)tcur : 0.f;
    for (int k = kL - 2; k >= 0; --k) {
        tcur = hist[(k + 1) * 16 + tcur];
        outp[k] = mk[k] ? (float)tcur : 0.f;
    }
}

__global__ void finalize_kernel(const float* __restrict__ llbuf, float* __restrict__ out) {
    if (threadIdx.x != 0) return;
    float p = 0.f, c = 0.f;
    for (int i = 0; i < kB; ++i) { p += llbuf[i]; c += llbuf[kB + i]; }
    out[0] = -(p + c); out[1] = p; out[2] = c;
}

// ---------------------------------------------------------------------------
// Host launcher
// ---------------------------------------------------------------------------
extern "C" void kernel_launch(void* const* d_in, const int* in_sizes, int n_in,
                              void* d_out, int out_size, void* d_ws, size_t ws_size,
                              hipStream_t stream) {
    const int*   input_ids  = (const int*)d_in[0];
    const int*   token_type = (const int*)d_in[1];
    const int*   valid_ids  = (const int*)d_in[2];
    const int*   plabels    = (const int*)d_in[3];
    const int*   clabels    = (const int*)d_in[4];
    const int*   label_mask = (const int*)d_in[5];
    const float* word_emb   = (const float*)d_in[6];
    const float* pos_emb    = (const float*)d_in[7];
    const float* type_emb   = (const float*)d_in[8];
    const float* ln_g       = (const float*)d_in[9];
    const float* ln_b       = (const float*)d_in[10];
    const float* Wih_f      = (const float*)d_in[11];
    const float* Whh_f      = (const float*)d_in[12];
    const float* bih        = (const float*)d_in[13];
    const float* bhh        = (const float*)d_in[14];
    const float* pW         = (const float*)d_in[15];
    const float* pb         = (const float*)d_in[16];
    const float* cW         = (const float*)d_in[17];
    const float* cb         = (const float*)d_in[18];
    const float* p_start    = (const float*)d_in[19];
    const float* p_end      = (const float*)d_in[20];
    const float* p_trans    = (const float*)d_in[21];
    const float* c_start    = (const float*)d_in[22];
    const float* c_end      = (const float*)d_in[23];
    const float* c_trans    = (const float*)d_in[24];

    // workspace carve-up (~104 MB total)
    char* base = (char*)d_ws;
    size_t off = 0;
    auto take = [&](size_t sz) -> void* {
        void* p = base + off;
        off = (off + sz + 255) & ~(size_t)255;
        return p;
    };
    unsigned short* xbf    = (unsigned short*)take((size_t)kB * kL * kD * 2);
    unsigned short* wih_bf = (unsigned short*)take((size_t)2 * 2 * kG * kD * 2);
    unsigned short* whh_bf = (unsigned short*)take((size_t)2 * 2 * kG * kH * 2);
    float*          biasws = (float*)take((size_t)2 * 2 * kG * 4);
    unsigned short* gpre   = (unsigned short*)take((size_t)2 * kB * kL * kG * 2);
    float*          hout   = (float*)take((size_t)kB * kL * kD * 4);
    float*          plog   = (float*)take((size_t)kB * kL * kNP * 4);
    float*          clog   = (float*)take((size_t)kB * kL * kNC * 4);
    int*            dest   = (int*)take((size_t)kB * kL * 4);
    int*            bp     = (int*)take((size_t)2 * kB * kL * 16 * 4);
    float*          llbuf  = (float*)take((size_t)2 * kB * 4);

    // 1) weights -> bf16, bias = bih + bhh
    {
        const int nWih = 2 * 2 * kG * kD;
        f32_to_bf16_kernel<<<(nWih + 255) / 256, 256, 0, stream>>>(Wih_f, wih_bf, nWih);
        const int nWhh = 2 * 2 * kG * kH;
        f32_to_bf16_kernel<<<(nWhh + 255) / 256, 256, 0, stream>>>(Whh_f, whh_bf, nWhh);
        const int nBias = 2 * 2 * kG;
        bias_sum_kernel<<<(nBias + 255) / 256, 256, 0, stream>>>(bih, bhh, biasws, nBias);
    }

    // 2) embedding + LN -> bf16 layer-0 input
    embed_ln_kernel<<<kB * kL, 256, 0, stream>>>(
        input_ids, token_type, word_emb, pos_emb, type_emb, ln_g, ln_b, xbf);

    // 3) two BiLSTM layers: big WMMA GEMM (input proj) + persistent recurrence
    for (int layer = 0; layer < 2; ++layer) {
        gemm_gates_kernel<<<dim3(kB * kL / 64, kG / 128, 2), 256, 0, stream>>>(
            xbf, wih_bf + (size_t)layer * 2 * kG * kD,
            biasws + (size_t)layer * 2 * kG, gpre);
        lstm_layer_kernel<<<2, 768, 0, stream>>>(
            gpre, whh_bf + (size_t)layer * 2 * kG * kH, hout);
        if (layer == 0) {
            const int n = kB * kL * kD;
            f32_to_bf16_kernel<<<(n + 255) / 256, 256, 0, stream>>>(hout, xbf, n);
        }
    }

    // 4) compaction + heads
    dest_kernel<<<1, 32, 0, stream>>>(valid_ids, dest);
    init_heads_kernel<<<kB * kL, 32, 0, stream>>>(pb, cb, plog, clog);
    logits_kernel<<<kB * kL, 384, 0, stream>>>(
        hout, valid_ids, dest, pW, pb, cW, cb, plog, clog);

    // 5) CRF ll + Viterbi, tags straight into d_out[3:]
    float* out_f = (float*)d_out;
    crf_kernel<<<dim3(kB, 2), 32, 0, stream>>>(
        plog, clog, plabels, clabels, label_mask,
        p_start, p_end, p_trans, c_start, c_end, c_trans,
        llbuf, out_f + 3, bp);
    finalize_kernel<<<1, 32, 0, stream>>>(llbuf, out_f);
    (void)in_sizes; (void)n_in; (void)out_size; (void)ws_size;
}
